// GraphMultiHeadAttention_78941498901013
// MI455X (gfx1250) — matrix operand order repaired and checked by reading, not compiled, not run on hardware
//
#include <hip/hip_runtime.h>
#include <hip/hip_bf16.h>

#define NUM_NODES  10240
#define NUM_GRAPHS 64
#define HIDDEN     512
#define HEAD_DIM   64
#define NODES_PAD  10272   // NUM_NODES + 32 pad rows (zeroed)
#define NPADV      11264   // >= sum(ceil16(n_g)) (<=11200) + slack, multiple of 32

typedef _Float16 h8  __attribute__((ext_vector_type(8)));
typedef _Float16 h16 __attribute__((ext_vector_type(16)));
typedef float    f8  __attribute__((ext_vector_type(8)));

// A-fragment (16x32 f16): lane m = lane&15, h = lane>>4.
// elements 0..15 = A[m][16h .. 16h+15]
__device__ __forceinline__ h16 load_a_frag(const _Float16* rowk, int h) {
    h8 lo = *(const h8*)(rowk + 16 * h);
    h8 hi = *(const h8*)(rowk + 16 * h + 8);
    return __builtin_shufflevector(lo, hi, 0,1,2,3,4,5,6,7,8,9,10,11,12,13,14,15);
}

// ---------------- metadata: starts / counts / padded vT bases ----------------
__global__ void meta_kernel(const int* __restrict__ batch, int* __restrict__ meta) {
    __shared__ int s[NUM_GRAPHS];
    const int t = threadIdx.x;
    if (t < NUM_GRAPHS) {
        int lo = 0, hi = NUM_NODES;
        while (lo < hi) { int mid = (lo + hi) >> 1; if (batch[mid] < t) lo = mid + 1; else hi = mid; }
        s[t] = lo;
    }
    __syncthreads();
    if (t == 0) {
        int p = 0;
        for (int g = 0; g < NUM_GRAPHS; ++g) {
            const int st  = s[g];
            const int en  = (g == NUM_GRAPHS - 1) ? NUM_NODES : s[g + 1];
            const int cnt = en - st;
            meta[g]       = st;       // start
            meta[64 + g]  = cnt;      // count
            meta[128 + g] = p;        // 16-aligned column base in vT
            p += (cnt + 15) & ~15;
        }
    }
}

// ---------------- f32 -> f16 conversions ----------------
__global__ void convert_x_kernel(const float* __restrict__ x, _Float16* __restrict__ xh) {
    const size_t N = (size_t)NUM_NODES * HIDDEN;
    for (size_t i = (size_t)blockIdx.x * blockDim.x + threadIdx.x; i < N;
         i += (size_t)gridDim.x * blockDim.x)
        xh[i] = (_Float16)x[i];
}

__global__ void convert_w_kernel(const float* __restrict__ wq, const float* __restrict__ wk,
                                 const float* __restrict__ wv, const float* __restrict__ wo,
                                 _Float16* __restrict__ wh) {
    const size_t M = (size_t)HIDDEN * HIDDEN;   // 262144, power of two
    for (size_t i = (size_t)blockIdx.x * blockDim.x + threadIdx.x; i < 4 * M;
         i += (size_t)gridDim.x * blockDim.x) {
        const float* src = (i < M) ? wq : (i < 2 * M) ? wk : (i < 3 * M) ? wv : wo;
        wh[i] = (_Float16)src[i & (M - 1)];
    }
}

// ---------------- zero-fill vT + pad rows of qh/kh ----------------
__global__ void zero_pad_kernel(_Float16* __restrict__ vT, _Float16* __restrict__ qh,
                                _Float16* __restrict__ kh) {
    const size_t VT  = (size_t)HIDDEN * NPADV;
    const size_t PAD = (size_t)(NODES_PAD - NUM_NODES) * HIDDEN;
    const size_t tot = VT + 2 * PAD;
    for (size_t i = (size_t)blockIdx.x * blockDim.x + threadIdx.x; i < tot;
         i += (size_t)gridDim.x * blockDim.x) {
        if (i < VT)            vT[i] = (_Float16)0.0f;
        else if (i < VT + PAD) qh[(size_t)NUM_NODES * HIDDEN + (i - VT)] = (_Float16)0.0f;
        else                   kh[(size_t)NUM_NODES * HIDDEN + (i - VT - PAD)] = (_Float16)0.0f;
    }
}

// ---------------- QKV projection: one wave = 32x32 tile (2x2 WMMA blocking) ----------------
__global__ __launch_bounds__(32) void gemm_qkv_kernel(
    const _Float16* __restrict__ xh, const _Float16* __restrict__ wh,
    const float* __restrict__ bq, const float* __restrict__ bk, const float* __restrict__ bv,
    const int* __restrict__ batch, const int* __restrict__ meta,
    _Float16* __restrict__ qh, _Float16* __restrict__ kh, _Float16* __restrict__ vT)
{
    const int lane = threadIdx.x;
    const int h = lane >> 4, n = lane & 15;
    const int m0 = blockIdx.x * 32;
    const int n0 = blockIdx.y * 32;
    const int sel = blockIdx.z;                      // 0=Q 1=K 2=V
    const _Float16* w = wh + (size_t)sel * HIDDEN * HIDDEN;
    const _Float16* arow[2] = { xh + (size_t)(m0 + n) * HIDDEN,
                                xh + (size_t)(m0 + 16 + n) * HIDDEN };
    const _Float16* brow[2] = { w  + (size_t)(n0 + n) * HIDDEN,
                                w  + (size_t)(n0 + 16 + n) * HIDDEN };
    f8 acc[2][2];
    f8 zero8 = {};
#pragma unroll
    for (int i = 0; i < 2; ++i)
#pragma unroll
        for (int j = 0; j < 2; ++j) acc[i][j] = zero8;

#pragma unroll 4
    for (int k0 = 0; k0 < HIDDEN; k0 += 32) {
        h16 a0 = load_a_frag(arow[0] + k0, h);
        h16 a1 = load_a_frag(arow[1] + k0, h);
        h16 b0 = *(const h16*)(brow[0] + k0 + 16 * h);
        h16 b1 = *(const h16*)(brow[1] + k0 + 16 * h);
        acc[0][0] = __builtin_amdgcn_wmma_f32_16x16x32_f16(false, a0, false, b0, (short)0, acc[0][0], false, false);
        acc[0][1] = __builtin_amdgcn_wmma_f32_16x16x32_f16(false, a0, false, b1, (short)0, acc[0][1], false, false);
        acc[1][0] = __builtin_amdgcn_wmma_f32_16x16x32_f16(false, a1, false, b0, (short)0, acc[1][0], false, false);
        acc[1][1] = __builtin_amdgcn_wmma_f32_16x16x32_f16(false, a1, false, b1, (short)0, acc[1][1], false, false);
    }

    const float* bias = (sel == 0) ? bq : ((sel == 1) ? bk : bv);
    const float bb[2] = { bias[n0 + n], bias[n0 + 16 + n] };

    if (sel == 0) {
#pragma unroll
        for (int i = 0; i < 2; ++i)
#pragma unroll
            for (int j = 0; j < 2; ++j)
#pragma unroll
                for (int r = 0; r < 8; ++r)   // fold softmax scale 1/sqrt(64) into Q
                    qh[(size_t)(m0 + 16 * i + r + 8 * h) * HIDDEN + n0 + 16 * j + n] =
                        (_Float16)((acc[i][j][r] + bb[j]) * 0.125f);
    } else if (sel == 1) {
#pragma unroll
        for (int i = 0; i < 2; ++i)
#pragma unroll
            for (int j = 0; j < 2; ++j)
#pragma unroll
                for (int r = 0; r < 8; ++r)
                    kh[(size_t)(m0 + 16 * i + r + 8 * h) * HIDDEN + n0 + 16 * j + n] =
                        (_Float16)(acc[i][j][r] + bb[j]);
    } else {
#pragma unroll
        for (int i = 0; i < 2; ++i)
#pragma unroll
            for (int r = 0; r < 8; ++r) {     // V stored transposed: vT[dim][graph-padded node]
                const int node = m0 + 16 * i + r + 8 * h;
                const int gg   = batch[node];
                const int col  = meta[128 + gg] + (node - meta[gg]);
#pragma unroll
                for (int j = 0; j < 2; ++j)
                    vT[(size_t)(n0 + 16 * j + n) * NPADV + col] = (_Float16)(acc[i][j][r] + bb[j]);
            }
    }
}

// ---------------- flash attention: one wave = (graph, head, 16-query tile) ----------------
__global__ __launch_bounds__(32) void attn_kernel(
    const _Float16* __restrict__ qh, const _Float16* __restrict__ kh,
    const _Float16* __restrict__ vT, const int* __restrict__ meta,
    _Float16* __restrict__ obuf)
{
    const int g = blockIdx.z, head = blockIdx.y, qt = blockIdx.x;
    const int start = meta[g], ng = meta[64 + g], pbase = meta[128 + g];
    if (qt * 16 >= ng) return;
    const int lane = threadIdx.x;
    const int h = lane >> 4, n = lane & 15;
    __shared__ alignas(16) _Float16 sP[16 * 32];

    const _Float16* qrow = qh + (size_t)(start + qt * 16 + n) * HIDDEN + head * HEAD_DIM;
    h16 qf0 = load_a_frag(qrow,      h);   // head-dim k = 0..31
    h16 qf1 = load_a_frag(qrow + 32, h);   // head-dim k = 32..63

    float mrow[8], lrow[8];
    f8 oacc[4];
    f8 zero8 = {};
#pragma unroll
    for (int r = 0; r < 8; ++r) { mrow[r] = -1e30f; lrow[r] = 0.0f; }
#pragma unroll
    for (int nt = 0; nt < 4; ++nt) oacc[nt] = zero8;

    for (int kb = 0; kb < ng; kb += 32) {
        // ---- S = Q K^T for 16x32 key block (two 16x16 column tiles) ----
        float sa[2][8];
#pragma unroll
        for (int ct = 0; ct < 2; ++ct) {
            const _Float16* krow = kh + (size_t)(start + kb + ct * 16 + n) * HIDDEN + head * HEAD_DIM;
            h16 kf0 = *(const h16*)(krow + 16 * h);
            h16 kf1 = *(const h16*)(krow + 32 + 16 * h);
            f8 cc = {};
            cc = __builtin_amdgcn_wmma_f32_16x16x32_f16(false, qf0, false, kf0, (short)0, cc, false, false);
            cc = __builtin_amdgcn_wmma_f32_16x16x32_f16(false, qf1, false, kf1, (short)0, cc, false, false);
            const bool valid = (kb + ct * 16 + n) < ng;   // mask padded keys
#pragma unroll
            for (int r = 0; r < 8; ++r) sa[ct][r] = valid ? cc[r] : -1e30f;
        }
        // ---- online softmax (rows r+8h live in the 16-lane half with same h) ----
        float rmax[8];
#pragma unroll
        for (int r = 0; r < 8; ++r) rmax[r] = fmaxf(sa[0][r], sa[1][r]);
#pragma unroll
        for (int off = 1; off < 16; off <<= 1)
#pragma unroll
            for (int r = 0; r < 8; ++r)
                rmax[r] = fmaxf(rmax[r], __shfl_xor(rmax[r], off, 32));

        float p0[8], p1[8], psum[8];
#pragma unroll
        for (int r = 0; r < 8; ++r) {
            const float mn    = fmaxf(mrow[r], rmax[r]);
            const float alpha = __expf(mrow[r] - mn);
            mrow[r] = mn;
            p0[r] = __expf(sa[0][r] - mn);
            p1[r] = __expf(sa[1][r] - mn);
            psum[r] = p0[r] + p1[r];
            lrow[r] *= alpha;
#pragma unroll
            for (int nt = 0; nt < 4; ++nt) oacc[nt][r] *= alpha;
        }
#pragma unroll
        for (int off = 1; off < 16; off <<= 1)
#pragma unroll
            for (int r = 0; r < 8; ++r)
                psum[r] += __shfl_xor(psum[r], off, 32);
#pragma unroll
        for (int r = 0; r < 8; ++r) lrow[r] += psum[r];

        // ---- P through LDS: C/D layout -> A layout ----
        __syncthreads();
#pragma unroll
        for (int r = 0; r < 8; ++r) {
            sP[(r + 8 * h) * 32 + n]      = (_Float16)p0[r];
            sP[(r + 8 * h) * 32 + 16 + n] = (_Float16)p1[r];
        }
        __syncthreads();
        h16 pf = load_a_frag(sP + n * 32, h);

        // ---- O += P V : 4 dim-tiles of 16, contraction over 32 keys ----
#pragma unroll
        for (int nt = 0; nt < 4; ++nt) {
            const _Float16* vrow = vT + (size_t)(head * HEAD_DIM + nt * 16 + n) * NPADV
                                      + pbase + kb + 16 * h;
            h16 vf = *(const h16*)vrow;
            oacc[nt] = __builtin_amdgcn_wmma_f32_16x16x32_f16(false, pf, false, vf, (short)0,
                                                              oacc[nt], false, false);
        }
    }
    // ---- normalize and store valid rows ----
#pragma unroll
    for (int r = 0; r < 8; ++r) {
        const int m = r + 8 * h;
        if (qt * 16 + m < ng) {
            const float inv = 1.0f / lrow[r];
            _Float16* orow = obuf + (size_t)(start + qt * 16 + m) * HIDDEN + head * HEAD_DIM;
#pragma unroll
            for (int nt = 0; nt < 4; ++nt)
                orow[nt * 16 + n] = (_Float16)(oacc[nt][r] * inv);
        }
    }
}

// ---------------- output projection: one wave = 32x32 tile (2x2 WMMA blocking) ----------------
__global__ __launch_bounds__(32) void gemm_out_kernel(
    const _Float16* __restrict__ oh, const _Float16* __restrict__ woh,
    const float* __restrict__ bo, float* __restrict__ out)
{
    const int lane = threadIdx.x;
    const int h = lane >> 4, n = lane & 15;
    const int m0 = blockIdx.x * 32;
    const int n0 = blockIdx.y * 32;
    const _Float16* arow[2] = { oh  + (size_t)(m0 + n) * HIDDEN,
                                oh  + (size_t)(m0 + 16 + n) * HIDDEN };
    const _Float16* brow[2] = { woh + (size_t)(n0 + n) * HIDDEN,
                                woh + (size_t)(n0 + 16 + n) * HIDDEN };
    f8 acc[2][2];
    f8 zero8 = {};
#pragma unroll
    for (int i = 0; i < 2; ++i)
#pragma unroll
        for (int j = 0; j < 2; ++j) acc[i][j] = zero8;

#pragma unroll 4
    for (int k0 = 0; k0 < HIDDEN; k0 += 32) {
        h16 a0 = load_a_frag(arow[0] + k0, h);
        h16 a1 = load_a_frag(arow[1] + k0, h);
        h16 b0 = *(const h16*)(brow[0] + k0 + 16 * h);
        h16 b1 = *(const h16*)(brow[1] + k0 + 16 * h);
        acc[0][0] = __builtin_amdgcn_wmma_f32_16x16x32_f16(false, a0, false, b0, (short)0, acc[0][0], false, false);
        acc[0][1] = __builtin_amdgcn_wmma_f32_16x16x32_f16(false, a0, false, b1, (short)0, acc[0][1], false, false);
        acc[1][0] = __builtin_amdgcn_wmma_f32_16x16x32_f16(false, a1, false, b0, (short)0, acc[1][0], false, false);
        acc[1][1] = __builtin_amdgcn_wmma_f32_16x16x32_f16(false, a1, false, b1, (short)0, acc[1][1], false, false);
    }
    const float bb[2] = { bo[n0 + n], bo[n0 + 16 + n] };
#pragma unroll
    for (int i = 0; i < 2; ++i)
#pragma unroll
        for (int j = 0; j < 2; ++j)
#pragma unroll
            for (int r = 0; r < 8; ++r)
                out[(size_t)(m0 + 16 * i + r + 8 * h) * HIDDEN + n0 + 16 * j + n] =
                    acc[i][j][r] + bb[j];
}

// ---------------- launch ----------------
extern "C" void kernel_launch(void* const* d_in, const int* in_sizes, int n_in,
                              void* d_out, int out_size, void* d_ws, size_t ws_size,
                              hipStream_t stream) {
    const float* x     = (const float*)d_in[0];
    const int*   batch = (const int*)  d_in[1];
    const float* wq    = (const float*)d_in[2];
    const float* bq    = (const float*)d_in[3];
    const float* wk    = (const float*)d_in[4];
    const float* bk    = (const float*)d_in[5];
    const float* wv    = (const float*)d_in[6];
    const float* bv    = (const float*)d_in[7];
    const float* wo    = (const float*)d_in[8];
    const float* bo    = (const float*)d_in[9];
    float* out = (float*)d_out;

    char* ws = (char*)d_ws;
    const size_t XH_OFF   = 0;                                            // 10,485,760 B (reused as attn out)
    const size_t WH_OFF   = XH_OFF + (size_t)NUM_NODES * HIDDEN * 2;      //  2,097,152 B
    const size_t QH_OFF   = WH_OFF + (size_t)4 * HIDDEN * HIDDEN * 2;     // 10,518,528 B
    const size_t KH_OFF   = QH_OFF + (size_t)NODES_PAD * HIDDEN * 2;      // 10,518,528 B
    const size_t VT_OFF   = KH_OFF + (size_t)NODES_PAD * HIDDEN * 2;      // 11,534,336 B
    const size_t META_OFF = VT_OFF + (size_t)HIDDEN * NPADV * 2;          //        768 B

    _Float16* xh   = (_Float16*)(ws + XH_OFF);   // also obuf
    _Float16* wh   = (_Float16*)(ws + WH_OFF);
    _Float16* qh   = (_Float16*)(ws + QH_OFF);
    _Float16* kh   = (_Float16*)(ws + KH_OFF);
    _Float16* vT   = (_Float16*)(ws + VT_OFF);
    int*      meta = (int*)     (ws + META_OFF);

    meta_kernel<<<1, 64, 0, stream>>>(batch, meta);
    convert_x_kernel<<<2048, 256, 0, stream>>>(x, xh);
    convert_w_kernel<<<1024, 256, 0, stream>>>(wq, wk, wv, wo, wh);
    zero_pad_kernel<<<2048, 256, 0, stream>>>(vT, qh, kh);

    gemm_qkv_kernel<<<dim3(NUM_NODES / 32, HIDDEN / 32, 3), 32, 0, stream>>>(
        xh, wh, bq, bk, bv, batch, meta, qh, kh, vT);

    attn_kernel<<<dim3(24 /*MAX_NODES/16*/, 8 /*heads*/, NUM_GRAPHS), 32, 0, stream>>>(
        qh, kh, vT, meta, xh /*obuf, xh no longer needed*/);

    gemm_out_kernel<<<dim3(NUM_NODES / 32, HIDDEN / 32), 32, 0, stream>>>(
        xh, wh + (size_t)3 * HIDDEN * HIDDEN, bo, out);
}
